// RebarGradient_4647154614849
// MI455X (gfx1250) — matrix operand order repaired and checked by reading, not compiled, not run on hardware
//
#include <hip/hip_runtime.h>
#include <hip/hip_bf16.h>
#include <stdint.h>

// Problem constants (from reference setup_inputs)
static constexpr int B_ = 4, S_ = 512, V_ = 32000, D_ = 256;
static constexpr int ROWS = B_ * S_;   // 2048
static constexpr int NITER = 3;
#define TAU_INV 1.0f
#define BIGV    1e10f
#define NEGV   -1e9f
#define EPSV    1e-20f

typedef __attribute__((ext_vector_type(4)))  float        f4;
typedef __attribute__((ext_vector_type(4)))  unsigned int u4v;
typedef __attribute__((ext_vector_type(16))) __bf16       v16bf;
typedef __attribute__((ext_vector_type(8)))  float        v8f;

union BFrag { v16bf v; u4v q[2]; };           // 32B WMMA A/B fragment
union Pack8 { u4v q; unsigned short u[8]; };  // 8 bf16 packed store

__device__ __forceinline__ float gumbelf(float u) {
    return fminf(-__logf(-__logf(u)), BIGV);
}
// round-to-nearest-even f32 -> bf16 bits
__device__ __forceinline__ unsigned short f2bf(float f) {
    unsigned u = __float_as_uint(f);
    u += 0x7FFFu + ((u >> 16) & 1u);
    return (unsigned short)(u >> 16);
}
__device__ __forceinline__ void onl(float& m, float& s, float z) {
    if (z > m) { s = s * __expf(m - z) + 1.f; m = z; }
    else       { s += __expf(z - m); }
}
__device__ __forceinline__ void onlArg(float& m, float& s, int& ai, float z, int i) {
    if (z > m) { s = s * __expf(m - z) + 1.f; m = z; ai = i; }
    else       { s += __expf(z - m); }
}

// ---------------------------------------------------------------------------
// Kernel B: emb f32 -> bf16, written in WMMA B-fragment order.
// Fragment f = (k5, nt): 512 contiguous bf16; element (lane,j):
//   k = k5*32 + (lane>>4)*16 + j ; n = nt*16 + (lane&15)
// => contiguous k order, same as the A frags (lanes 0-15: K 0..15, lanes 16-31: K 16..31),
//    each lane's 16 values contiguous (32B) for ds_load_b128 pairs.
// ---------------------------------------------------------------------------
__global__ void emb_swizzle_bf16(const float* __restrict__ emb,
                                 unsigned short* __restrict__ embb) {
    int o = blockIdx.x * blockDim.x + threadIdx.x;
    if (o >= V_ * D_) return;
    int f    = o >> 9;
    int e    = o & 511;
    int lane = e >> 4, j = e & 15;
    int k5 = f >> 4, nt = f & 15;
    int k = k5 * 32 + ((lane >> 4) << 4) + j;
    int n = (nt << 4) + (lane & 15);
    embb[o] = f2bf(emb[(size_t)k * D_ + n]);
}

// ---------------------------------------------------------------------------
// Kernel A: one block (512 threads) per (b,s) row; 260 KB LDS working set
// (logits row + scratch row), float4 streaming, NT hints on single-use data.
// ---------------------------------------------------------------------------
__global__ __launch_bounds__(512) void row_pipeline(
    const float* __restrict__ logits, const float* __restrict__ emb,
    const float* __restrict__ u_g1,   const float* __restrict__ u_t0,
    const float* __restrict__ u_other,const float* __restrict__ u_target,
    float* __restrict__ out_f, float* __restrict__ out_p1,
    float* __restrict__ out_hard, float* __restrict__ stats)
{
    extern __shared__ char smem[];
    float* Ls  = (float*)smem;                              // V floats
    float* Zs  = (float*)(smem + (size_t)V_ * 4);           // V floats
    float* red = (float*)(smem + (size_t)V_ * 8);           // 512 floats
    int*  redi = (int*)  (smem + (size_t)V_ * 8 + 2048);    // 512 ints
    float* bc  = (float*)(smem + (size_t)V_ * 8 + 4096);    // 8 broadcast scalars

    const int tid = threadIdx.x, nt = blockDim.x;
    const int row = blockIdx.x;
    const int nv4 = V_ / 4;                                 // 8000
    const f4* lrow4 = (const f4*)(logits + (size_t)row * V_);
    const f4* urow4 = (const f4*)(u_g1   + (size_t)row * V_);

    // ---- P1: stage logits/z to LDS; online max + argmax + sum(exp) ----
    float m = -3.4e38f, s = 0.f; int ai = 0;
    for (int v4 = tid; v4 < nv4; v4 += nt) {
        f4 lg = lrow4[v4];
        f4 u  = __builtin_nontemporal_load(urow4 + v4);   // single-use stream
        ((f4*)Ls)[v4] = lg;
        f4 z;
        z.x = (lg.x + gumbelf(u.x)) * TAU_INV;
        z.y = (lg.y + gumbelf(u.y)) * TAU_INV;
        z.z = (lg.z + gumbelf(u.z)) * TAU_INV;
        z.w = (lg.w + gumbelf(u.w)) * TAU_INV;
        ((f4*)Zs)[v4] = z;
        int base = v4 * 4;
        onlArg(m, s, ai, z.x, base + 0);
        onlArg(m, s, ai, z.y, base + 1);
        onlArg(m, s, ai, z.z, base + 2);
        onlArg(m, s, ai, z.w, base + 3);
    }
    red[tid] = m; redi[tid] = ai;
    __syncthreads();
    for (int off = nt >> 1; off > 0; off >>= 1) {
        if (tid < off) {
            float o = red[tid + off]; int oi = redi[tid + off];
            if (o > red[tid] || (o == red[tid] && oi < redi[tid])) { red[tid] = o; redi[tid] = oi; }
        }
        __syncthreads();
    }
    const float M1 = red[0];
    const int   idx = redi[0];
    __syncthreads();
    red[tid] = s * __expf(m - M1);
    __syncthreads();
    for (int off = nt >> 1; off > 0; off >>= 1) {
        if (tid < off) red[tid] += red[tid + off];
        __syncthreads();
    }
    if (tid == 0) {
        bc[0] = M1 + __logf(red[0]);     // log-normalizer for p1
        bc[2] = gumbelf(u_t0[row]);      // g_t carry init
    }
    __syncthreads();
    const float tl    = Ls[idx];         // target_logits (LDS broadcast)
    const float norm1 = bc[0];

    // ---- P2: write p1 / hard (NT stores: outputs never re-read) ----
    f4* p1row = (f4*)(out_p1  + (size_t)row * V_);
    f4* hdrow = (f4*)(out_hard + (size_t)row * V_);
    for (int v4 = tid; v4 < nv4; v4 += nt) {
        f4 z = ((const f4*)Zs)[v4];
        f4 p;
        p.x = __expf(z.x - norm1); p.y = __expf(z.y - norm1);
        p.z = __expf(z.z - norm1); p.w = __expf(z.w - norm1);
        __builtin_nontemporal_store(p, p1row + v4);
        int base = v4 * 4;
        f4 h;
        h.x = (base + 0 == idx) ? 1.f : 0.f;
        h.y = (base + 1 == idx) ? 1.f : 0.f;
        h.z = (base + 2 == idx) ? 1.f : 0.f;
        h.w = (base + 3 == idx) ? 1.f : 0.f;
        __builtin_nontemporal_store(h, hdrow + v4);
    }
    if (tid < D_) out_f[(size_t)row * D_ + tid] = emb[(size_t)idx * D_ + tid];

    // ---- scan: 3 conditional-Gumbel iterations; stash last g_o in Zs ----
    for (int it = 0; it < NITER; ++it) {
        float gt = bc[2];
        if (it == NITER - 1 && tid == 0) bc[3] = gt;   // carry entering last iter
        const f4* uo4 = (const f4*)(u_other + ((size_t)it * ROWS + row) * (size_t)V_);
        float lmax = -3.4e38f;
        for (int v4 = tid; v4 < nv4; v4 += nt) {
            f4 lg = ((const f4*)Ls)[v4];
            // iterations 0,1 are single-use; iteration 2 is re-read by the GEMM kernel
            f4 uo = (it != NITER - 1) ? __builtin_nontemporal_load(uo4 + v4) : uo4[v4];
            f4 go;
            go.x = -__logf(-__logf(uo.x * __expf(-__expf(-(gt + tl - lg.x))) + EPSV) + EPSV);
            go.y = -__logf(-__logf(uo.y * __expf(-__expf(-(gt + tl - lg.y))) + EPSV) + EPSV);
            go.z = -__logf(-__logf(uo.z * __expf(-__expf(-(gt + tl - lg.z))) + EPSV) + EPSV);
            go.w = -__logf(-__logf(uo.w * __expf(-__expf(-(gt + tl - lg.w))) + EPSV) + EPSV);
            if (it == NITER - 1) ((f4*)Zs)[v4] = go;
            int base = v4 * 4;
            lmax = fmaxf(lmax, (base + 0 == idx) ? NEGV : (go.x + lg.x - tl));
            lmax = fmaxf(lmax, (base + 1 == idx) ? NEGV : (go.y + lg.y - tl));
            lmax = fmaxf(lmax, (base + 2 == idx) ? NEGV : (go.z + lg.z - tl));
            lmax = fmaxf(lmax, (base + 3 == idx) ? NEGV : (go.w + lg.w - tl));
        }
        red[tid] = lmax;
        __syncthreads();
        for (int off = nt >> 1; off > 0; off >>= 1) {
            if (tid < off) red[tid] = fmaxf(red[tid], red[tid + off]);
            __syncthreads();
        }
        if (tid == 0) {
            float low = __expf(-__expf(-red[0]));
            float ut  = u_target[(size_t)it * ROWS + row];
            bc[2] = -__logf(-__logf(ut * (1.f - low) + low + EPSV) + EPSV);
        }
        __syncthreads();
    }
    const float g_target = bc[2];
    const float gt2      = bc[3];

    // ---- P6: softmax stats for p2 (LDS only) ----
    float m2 = -3.4e38f, s2 = 0.f;
    for (int v4 = tid; v4 < nv4; v4 += nt) {
        f4 lg = ((const f4*)Ls)[v4];
        f4 go = ((const f4*)Zs)[v4];
        int base = v4 * 4;
        onl(m2, s2, (lg.x + ((base + 0 == idx) ? g_target : go.x)) * TAU_INV);
        onl(m2, s2, (lg.y + ((base + 1 == idx) ? g_target : go.y)) * TAU_INV);
        onl(m2, s2, (lg.z + ((base + 2 == idx) ? g_target : go.z)) * TAU_INV);
        onl(m2, s2, (lg.w + ((base + 3 == idx) ? g_target : go.w)) * TAU_INV);
    }
    red[tid] = m2;
    __syncthreads();
    for (int off = nt >> 1; off > 0; off >>= 1) {
        if (tid < off) red[tid] = fmaxf(red[tid], red[tid + off]);
        __syncthreads();
    }
    const float M2 = red[0];
    __syncthreads();
    red[tid] = s2 * __expf(m2 - M2);
    __syncthreads();
    for (int off = nt >> 1; off > 0; off >>= 1) {
        if (tid < off) red[tid] += red[tid + off];
        __syncthreads();
    }
    if (tid == 0) {
        float* st = stats + (size_t)row * 8;
        st[0] = gt2; st[1] = g_target; st[2] = tl;
        st[3] = M2;  st[4] = 1.f / red[0];
        st[5] = __int_as_float(idx);
    }
}

// ---------------------------------------------------------------------------
// Kernel C: g = p2 @ emb via v_wmma_f32_16x16x32_bf16.
// Block = 16 rows, 256 threads = 8 waves; wave w owns N-tiles {2w, 2w+1}.
// Double-buffered async-to-LDS B staging (16 b128/wave/chunk):
//   issue chunk i+1, s_wait_asynccnt 0x10 retires chunk i -> DMA overlaps the
//   transcendental p2 recompute. Fragments are contiguous: 2x ds_load_b128.
// ---------------------------------------------------------------------------
static constexpr int A_STRIDE = 136;   // ushorts; 272B row stride (16B aligned, bank-spread)

__global__ __launch_bounds__(256) void wmma_gemm_p2(
    const float* __restrict__ logits, const float* __restrict__ u_other,
    const unsigned short* __restrict__ embb, const float* __restrict__ stats,
    float* __restrict__ out_g)
{
    extern __shared__ char smem[];
    unsigned short* At  = (unsigned short*)smem;                           // 16 x 136
    unsigned short* Bt0 = (unsigned short*)(smem + 16 * A_STRIDE * 2);     // 64 KB
    unsigned short* Bt1 = (unsigned short*)(smem + 16 * A_STRIDE * 2 + 65536);
    float* st = (float*)(smem + 16 * A_STRIDE * 2 + 2 * 65536);            // 16 x 8

    const int tid  = threadIdx.x;
    const int lane = tid & 31, wid = tid >> 5;
    const int r0   = blockIdx.x * 16;
    const int col  = lane & 15;        // A: row m; B/D: column within tile
    const int half = lane >> 4;

    if (tid < 128) st[tid] = stats[(size_t)(r0 + (tid >> 3)) * 8 + (tid & 7)];
    __syncthreads();

    // this thread computes row i, k-span [kk0, kk0+8) of the A tile
    const int ti  = tid >> 4;
    const int kk0 = (tid & 15) * 8;
    const float* rs = st + ti * 8;

    const unsigned ldsB[2] = { (unsigned)(uintptr_t)Bt0, (unsigned)(uintptr_t)Bt1 };

    // 64KB chunk [kc, kc+128) of swizzled embb is contiguous at element kc*256
    auto issueB = [&](int buf, int kc) {
        const unsigned short* src = embb + (size_t)kc * 256;
        unsigned base = ldsB[buf];
        #pragma unroll 4
        for (int t = tid; t < 4096; t += 256) {           // 16 x b128 per thread
            unsigned loff = base + (unsigned)t * 16u;
            const void* gaddr = (const void*)(src + (size_t)t * 8);
            asm volatile("global_load_async_to_lds_b128 %0, %1, off"
                         :: "v"(loff), "v"(gaddr) : "memory");
        }
    };

    v8f acc0 = {0,0,0,0,0,0,0,0};
    v8f acc1 = {0,0,0,0,0,0,0,0};

    issueB(0, 0);
    int cur = 0;
    for (int kc = 0; kc < V_; kc += 128) {
        // -- recompute p2 tile into At (overlaps in-flight async DMA) --
        {
            float gt2 = rs[0], g_t = rs[1], tl = rs[2], m2 = rs[3], inv_s2 = rs[4];
            int idx = __float_as_int(rs[5]);
            int r = r0 + ti;
            const f4* lp = (const f4*)(logits + (size_t)r * V_ + kc + kk0);
            const f4* up = (const f4*)(u_other + ((size_t)(NITER - 1) * ROWS + r) * (size_t)V_ + kc + kk0);
            f4 la = lp[0], lb = lp[1];
            f4 ua = __builtin_nontemporal_load(up);        // final use of u_other[2]
            f4 ub = __builtin_nontemporal_load(up + 1);
            float lgv[8] = {la.x, la.y, la.z, la.w, lb.x, lb.y, lb.z, lb.w};
            float uv[8]  = {ua.x, ua.y, ua.z, ua.w, ub.x, ub.y, ub.z, ub.w};
            Pack8 pk;
            #pragma unroll
            for (int c = 0; c < 8; ++c) {
                int k = kc + kk0 + c;
                float lg = lgv[c];
                float noise;
                if (k == idx) noise = g_t;
                else {
                    float upc = __expf(-__expf(-(gt2 + tl - lg)));
                    noise = -__logf(-__logf(uv[c] * upc + EPSV) + EPSV);
                }
                pk.u[c] = f2bf(__expf((lg + noise) * TAU_INV - m2) * inv_s2);
            }
            *(u4v*)(At + ti * A_STRIDE + kk0) = pk.q;
        }
        // -- prefetch next chunk into the other buffer, retire current --
        int nkc = (kc + 128 < V_) ? kc + 128 : 0;          // tail: harmless refetch
        issueB(cur ^ 1, nkc);
        asm volatile("s_wait_asynccnt 0x10" ::: "memory"); // oldest 16 (cur) done
        __syncthreads();

        const unsigned short* Bt = (cur == 0) ? Bt0 : Bt1;
        #pragma unroll
        for (int ks = 0; ks < 128; ks += 32) {
            // A frag: lane<16 -> K 0..15 ; lane>=16 -> K 16..31 (2x b128)
            BFrag a;
            const unsigned short* ap = At + col * A_STRIDE + ks + half * 16;
            a.q[0] = *(const u4v*)ap;
            a.q[1] = *(const u4v*)(ap + 8);
            // B frags: swizzled fragment = 512 contiguous bf16, lane slice = 32B
            BFrag b0, b1;
            const unsigned short* bp =
                Bt + (((ks >> 5) * 16 + wid * 2) * 512) + lane * 16;
            b0.q[0] = *(const u4v*)bp;
            b0.q[1] = *(const u4v*)(bp + 8);
            b1.q[0] = *(const u4v*)(bp + 512);
            b1.q[1] = *(const u4v*)(bp + 520);
            acc0 = __builtin_amdgcn_wmma_f32_16x16x32_bf16(false, a.v, false, b0.v,
                                                           (short)0, acc0, false, false);
            acc1 = __builtin_amdgcn_wmma_f32_16x16x32_bf16(false, a.v, false, b1.v,
                                                           (short)0, acc1, false, false);
        }
        __syncthreads();
        cur ^= 1;
    }

    // D layout: VGPR r -> M = r (lanes 0-15) / M = r+8 (lanes 16-31); N = lane&15
    const int n0 = wid * 32;
    #pragma unroll
    for (int rr = 0; rr < 8; ++rr) {
        int r = r0 + rr + half * 8;
        out_g[(size_t)r * D_ + n0 + col]      = acc0[rr];
        out_g[(size_t)r * D_ + n0 + 16 + col] = acc1[rr];
    }
}

// ---------------------------------------------------------------------------
extern "C" void kernel_launch(void* const* d_in, const int* in_sizes, int n_in,
                              void* d_out, int out_size, void* d_ws, size_t ws_size,
                              hipStream_t stream)
{
    const float* logits   = (const float*)d_in[0];
    const float* emb      = (const float*)d_in[1];
    // d_in[2..6] (sent_mask, f_mean, g_mean, g2_mean, fg_mean) unused by outputs
    const float* u_g1     = (const float*)d_in[7];
    const float* u_t0     = (const float*)d_in[8];
    const float* u_other  = (const float*)d_in[9];
    const float* u_target = (const float*)d_in[10];

    float* out      = (float*)d_out;
    float* out_f    = out;                                    // [2048,256]
    float* out_p1   = out_f  + (size_t)ROWS * D_;             // [2048,32000]
    float* out_hard = out_p1 + (size_t)ROWS * V_;             // [2048,32000]
    float* out_g    = out_hard + (size_t)ROWS * V_;           // [2048,256]

    unsigned short* embb = (unsigned short*)d_ws;                       // 16 MB (swizzled)
    float* stats = (float*)((char*)d_ws + (size_t)V_ * D_ * 2);         // 64 KB

    int nEmb = V_ * D_;
    emb_swizzle_bf16<<<(nEmb + 255) / 256, 256, 0, stream>>>(emb, embb);

    size_t shA = (size_t)V_ * 8 + 4096 + 64;   // 260160 B (< 320 KB WGP LDS)
    row_pipeline<<<ROWS, 512, shA, stream>>>(logits, emb, u_g1, u_t0, u_other, u_target,
                                             out_f, out_p1, out_hard, stats);

    size_t shC = 16 * A_STRIDE * 2 + 2 * 65536 + 16 * 8 * 4;  // ~136 KB
    wmma_gemm_p2<<<ROWS / 16, 256, shC, stream>>>(logits, u_other, embb, stats, out_g);
}
